// LearnableSparseHandler_12094627905783
// MI455X (gfx1250) — matrix operand order repaired and checked by reading, not compiled, not run on hardware
//
#include <hip/hip_runtime.h>
#include <hip/hip_bf16.h>
#include <math.h>

typedef __attribute__((ext_vector_type(16))) _Float16 v16h;
typedef __attribute__((ext_vector_type(8)))  float    v8f;
typedef __attribute__((ext_vector_type(4)))  float    v4f;

#define B_   8
#define T_   8
#define C_   64
#define H_   64
#define W_   64
#define N_   (H_*W_)          // 4096 spatial positions
#define KEEP 2048             // top-k per batch
#define OC   16               // conv1 out channels
#define KTOT 576              // 64 in-ch * 9 taps
#define NCHUNK 18             // 576 / 32 (K per WMMA)
#define TW   66               // padded tile width (x = -1 .. 64)

// ---------------------------------------------------------------------------
// Kernel 0: swizzle conv1 weights (f32 [16][64][3][3]) into the exact CDNA5
// 16-bit A-matrix (16x32) VGPR layout per K-chunk, stored f16 so each conv
// wave loads its A fragment as one contiguous v16h (32B) per chunk.
//   lane = (M | khalf<<4); element e -> Klocal = (e<8 ? 8*khalf+e : 16+8*khalf+e-8)
//   global K = chunk*32 + Klocal; K ordering: k = c*9 + kh*3 + kw
// ---------------------------------------------------------------------------
__global__ void prep_weights_kernel(const float* __restrict__ w1,
                                    _Float16* __restrict__ Aws) {
    int tid = blockIdx.x * blockDim.x + threadIdx.x;     // 0..9215 (= 18*32*16)
    int e     = tid & 15;
    int lane  = (tid >> 4) & 31;
    int chunk = tid >> 9;
    int m     = lane & 15;
    int khalf = lane >> 4;
    int kl = (e < 8) ? (khalf * 8 + e) : (16 + khalf * 8 + (e - 8));
    int k  = chunk * 32 + kl;                            // 0..575
    int c  = k / 9;
    int r  = k - c * 9;
    float v = w1[((m * C_ + c) * 3 + (r / 3)) * 3 + (r % 3)];
    Aws[tid] = (_Float16)v;                              // tid == (chunk*32+lane)*16+e
}

// ---------------------------------------------------------------------------
// Kernel 1: time-mean over T, f32 -> f16  (streaming read of x, 64 MiB)
// xm[b][c][h][w] = (1/8) * sum_t x[b][t][c][h][w]
// ---------------------------------------------------------------------------
__global__ void mean_kernel(const float* __restrict__ x,
                            _Float16* __restrict__ xm) {
    int tid = blockIdx.x * blockDim.x + threadIdx.x;     // 0..2097151
    int b   = tid >> 18;                                 // / 262144
    int chw = tid & 262143;
    const float* p = x + (size_t)b * (T_ * C_ * H_ * W_) + chw;
    float s = 0.f;
    #pragma unroll
    for (int t = 0; t < T_; ++t)
        s += __builtin_nontemporal_load(p + t * (C_ * H_ * W_));
    xm[tid] = (_Float16)(s * 0.125f);
}

// ---------------------------------------------------------------------------
// Kernel 2: router conv via LDS-staged WMMA implicit GEMM.
// One 128-thread block (4 waves) per (batch, output row y).
//  - Block stages xm rows y-1..y+1, all 64 channels, zero-padded in x, into
//    LDS tile[64][3][66] (25 KB) with branchless-divergence contiguous fills.
//  - Wave w computes pixels x0=w*16..+15: M=16 out-channels, K=576 in 18
//    chunks of 32; A from pre-swizzled global, B gathered from LDS (ds loads,
//    no bounds checks).  Then bias + leaky_relu + 1x1 conv + sigmoid.
// ---------------------------------------------------------------------------
__global__ void conv_wmma_kernel(const _Float16* __restrict__ xm,
                                 const _Float16* __restrict__ Aws,
                                 const float* __restrict__ b1,
                                 const float* __restrict__ w2,
                                 const float* __restrict__ b2,
                                 float* __restrict__ scores) {
    __shared__ _Float16 tile[C_ * 3 * TW];               // [c][kh][x+1], 25344 B

    int b = blockIdx.x >> 6;                             // 8 batches
    int y = blockIdx.x & 63;                             // 64 rows

    // ---- cooperative fill: 12672 halves / 128 threads = 99 each ----
    for (int idx = threadIdx.x; idx < C_ * 3 * TW; idx += 128) {
        int c   = idx / (3 * TW);
        int rem = idx - c * (3 * TW);
        int r   = rem / TW;                              // kh
        int xx  = rem - r * TW;                          // padded x index
        int yy  = y + r - 1;
        int xg  = xx - 1;
        _Float16 v = (_Float16)0.f;
        if (yy >= 0 && yy < H_ && xg >= 0 && xg < W_)
            v = xm[((b * C_ + c) * H_ + yy) * W_ + xg];
        tile[idx] = v;
    }
    __syncthreads();

    int lane  = threadIdx.x & 31;                        // EXEC all-1s per wave
    int wave  = threadIdx.x >> 5;
    int x0    = wave << 4;                               // 16 pixels per wave
    int n     = lane & 15;                               // B/C/D column
    int khalf = lane >> 4;

    v8f acc = {};
    for (int ch = 0; ch < NCHUNK; ++ch) {
        // A fragment: pre-swizzled, contiguous 32B per lane (L1/L2 resident)
        v16h a = *(const v16h*)(Aws + (ch * 32 + lane) * 16);
        // B fragment (32x16 f16): lanes 0-15 K=0..7,16..23, lanes 16-31 K=8..15,24..31
        v16h bf;
        #pragma unroll
        for (int e = 0; e < 16; ++e) {
            int k  = ch * 32 + ((e < 8) ? (khalf * 8 + e) : (16 + khalf * 8 + (e - 8)));
            int c  = k / 9;
            int r  = k - c * 9;
            int kh = r / 3;
            int kw = r - kh * 3;
            bf[e] = tile[(c * 3 + kh) * TW + x0 + n + kw];   // branchless ds load
        }
        acc = __builtin_amdgcn_wmma_f32_16x16x32_f16(
            /*neg_a=*/false, a, /*neg_b=*/false, bf,
            /*c_mod=*/(short)0, acc, /*reuse_a=*/false, /*reuse_b=*/false);
    }

    // C/D layout: VGPR r holds M = r (lanes 0-15) or r+8 (lanes 16-31), N = lane&15.
    float partial = 0.f;
    #pragma unroll
    for (int r = 0; r < 8; ++r) {
        int chn = khalf * 8 + r;
        float h = acc[r] + b1[chn];
        h = (h > 0.f) ? h : 0.01f * h;
        partial += w2[chn] * h;
    }
    partial += __shfl_xor(partial, 16, 32);              // combine channel halves
    if (lane < 16) {
        float z = partial + b2[0];
        float s = 1.f / (1.f + expf(-z));
        scores[b * N_ + y * W_ + x0 + n] = s;
    }
}

// ---------------------------------------------------------------------------
// Kernel 3: per-batch top-KEEP threshold via 32-round greedy radix select on
// uint bit patterns (sigmoid scores > 0, so uint order == float order), then
// keepscale[b][p] = (bits >= thr) ? s/(s+eps) : 0.   One block per batch.
// ---------------------------------------------------------------------------
__global__ void select_kernel(const float* __restrict__ scores,
                              float* __restrict__ keeps) {
    __shared__ unsigned sb[N_];
    __shared__ int cnt;
    int b   = blockIdx.x;
    int tid = threadIdx.x;

    for (int i = tid; i < N_; i += blockDim.x)
        sb[i] = __float_as_uint(scores[b * N_ + i]);
    __syncthreads();

    unsigned prefix = 0u;
    for (int bit = 31; bit >= 0; --bit) {
        unsigned cand = prefix | (1u << bit);
        if (tid == 0) cnt = 0;
        __syncthreads();
        int local = 0;
        for (int i = tid; i < N_; i += blockDim.x)
            local += (sb[i] >= cand) ? 1 : 0;
        atomicAdd(&cnt, local);
        __syncthreads();
        if (cnt >= KEEP) prefix = cand;   // uniform across block (cnt is shared)
        __syncthreads();
    }

    for (int i = tid; i < N_; i += blockDim.x) {
        unsigned u = sb[i];
        float s = __uint_as_float(u);
        keeps[b * N_ + i] = (u >= prefix) ? (s / (s + 1e-6f)) : 0.f;
    }
}

// ---------------------------------------------------------------------------
// Kernel 4: streaming masked multiply (the bandwidth-dominant pass).
// out[b,t,c,h,w] = x[b,t,c,h,w] * keepscale[b][h*W+w].  float4 vectorized,
// nontemporal on the 64 MiB x/out streams; keepscale stays hot in L2.
// ---------------------------------------------------------------------------
__global__ void mask_kernel(const float* __restrict__ x,
                            const float* __restrict__ keeps,
                            float* __restrict__ out) {
    int i4 = blockIdx.x * blockDim.x + threadIdx.x;      // 0..4194303
    int f  = i4 << 2;                                    // flat float index
    int p  = f & (N_ - 1);                               // spatial position
    int b  = f >> 21;                                    // / (T_*C_*H_*W_)
    v4f xv = __builtin_nontemporal_load((const v4f*)(x + f));
    v4f sv = *(const v4f*)(keeps + b * N_ + p);
    v4f r  = xv * sv;
    __builtin_nontemporal_store(r, (v4f*)(out + f));
}

// ---------------------------------------------------------------------------
extern "C" void kernel_launch(void* const* d_in, const int* in_sizes, int n_in,
                              void* d_out, int out_size, void* d_ws, size_t ws_size,
                              hipStream_t stream) {
    const float* x  = (const float*)d_in[0];   // [8,8,64,64,64]
    const float* w1 = (const float*)d_in[1];   // [16,64,3,3]
    const float* b1 = (const float*)d_in[2];   // [16]
    const float* w2 = (const float*)d_in[3];   // [1,16,1,1]
    const float* b2 = (const float*)d_in[4];   // [1]
    float* out = (float*)d_out;                // [8,8,64,64,64]

    char* ws = (char*)d_ws;
    _Float16* Aws    = (_Float16*)(ws);                         //  18 KB (rounded 32 KB)
    _Float16* xm     = (_Float16*)(ws + 32768);                 //   4 MiB f16 mean
    float*    scores = (float*)(ws + 32768 + 4194304);          // 128 KB
    float*    keeps  = (float*)(ws + 32768 + 4194304 + 131072); // 128 KB

    prep_weights_kernel<<<36,    256, 0, stream>>>(w1, Aws);            // 9216 elems
    mean_kernel        <<<8192,  256, 0, stream>>>(x, xm);              // 2,097,152 elems
    conv_wmma_kernel   <<<512,   128, 0, stream>>>(xm, Aws, b1, w2, b2, scores); // (b,y) blocks
    select_kernel      <<<B_,    256, 0, stream>>>(scores, keeps);
    mask_kernel        <<<16384, 256, 0, stream>>>(x, keeps, out);      // 4,194,304 float4
}
